// ClusteringLayer_4020089389723
// MI455X (gfx1250) — matrix operand order repaired and checked by reading, not compiled, not run on hardware
//
#include <hip/hip_runtime.h>

// ---------------------------------------------------------------------------
// ClusteringLayer on MI455X (gfx1250, wave32, WMMA).
// distances = ||x||^2 + ||c||^2 - 2*x@c^T via split-bf16 WMMA (xh*ch+xh*cl+xl*ch)
// then Student-t q, row-normalize (output 0), column sums, q^2 sharpening,
// argmax+threshold, scatter-accumulated cluster update (output 1).
// Round 2: wave tile 32x128 (B-fragment reuse, -41% LDS reads) + async
// global->LDS staging of cluster tiles (ASYNCcnt path).
// ---------------------------------------------------------------------------

#define N_ROWS 65536
#define DIM    1024
#define KC     256
#define THRESH 0.1f
#define EPS_F  2.2204460492503131e-16f   // np.finfo(float64).eps

typedef __attribute__((ext_vector_type(16))) __bf16 v16bf;
typedef __attribute__((ext_vector_type(8)))  __bf16 v8bf;
typedef __attribute__((ext_vector_type(8)))  float  v8f;
typedef __attribute__((address_space(3)))    char   lds_char;

// workspace layout (bytes)
static constexpr size_t WS_CSQ   = 0;                      // 256 f32
static constexpr size_t WS_QSUM  = 1024;                   // 256 f32
static constexpr size_t WS_DEN   = 2048;                   // 256 f32
static constexpr size_t WS_NUM   = 4096;                   // 256*1024 f32 (1 MB)
static constexpr size_t WS_CH    = WS_NUM   + 1048576;     // 256*1024 bf16 (512 KB)
static constexpr size_t WS_CL    = WS_CH    + 524288;      // 256*1024 bf16 (512 KB)
static constexpr size_t WS_QPART = WS_CL    + 524288;      // 512*256 f32 (512 KB)

// LDS offset of a generic pointer into a __shared__ array (for async-to-LDS VDST)
__device__ __forceinline__ uint32_t lds_offset_of(const void* p) {
    return (uint32_t)(uintptr_t)(lds_char*)p;
}
// Async 16B global->LDS copy, GVS mode (64-bit SGPR base + 32-bit VGPR offset).
__device__ __forceinline__ void async_copy_b128(uint32_t lds_off, const void* sbase, uint32_t voff) {
    asm volatile("global_load_async_to_lds_b128 %0, %1, %2"
                 :: "v"(lds_off), "v"(voff), "s"(sbase) : "memory");
}
__device__ __forceinline__ void wait_asynccnt0() {
    asm volatile("s_wait_asynccnt 0x0" ::: "memory");
}

// ---------------------------------------------------------------------------
__global__ void zero_ws_kernel(float* __restrict__ num, float* __restrict__ den) {
    int i = blockIdx.x * 256 + threadIdx.x;      // grid 1024 -> exactly 256*1024
    num[i] = 0.0f;
    if (blockIdx.x == 0) den[threadIdx.x] = 0.0f;
}

// split clusters into bf16 hi/lo and compute ||c||^2 (fp32-exact)
__global__ void prep_clusters_kernel(const float* __restrict__ clusters,
                                     __bf16* __restrict__ ch, __bf16* __restrict__ cl,
                                     float* __restrict__ csq) {
    const int k = blockIdx.x, t = threadIdx.x;
    float s = 0.0f;
    for (int i = t; i < DIM; i += 256) {
        float v  = clusters[(size_t)k * DIM + i];
        __bf16 h = (__bf16)v;
        __bf16 l = (__bf16)(v - (float)h);
        ch[(size_t)k * DIM + i] = h;
        cl[(size_t)k * DIM + i] = l;
        s += v * v;
    }
    __shared__ float red[256];
    red[t] = s; __syncthreads();
    for (int o = 128; o > 0; o >>= 1) { if (t < o) red[t] += red[t + o]; __syncthreads(); }
    if (t == 0) csq[k] = red[0];
}

// ---------------------------------------------------------------------------
// Main pass: 512 blocks x 256 threads. Block = 128 rows x 256 cols.
// Wave w: row group rw = w&3 (rows rw*32..+31), col half cw = w>>2 (cols cw*128..+127).
// ---------------------------------------------------------------------------
__global__ __launch_bounds__(256)
void pass1_kernel(const float* __restrict__ x,
                  const __bf16* __restrict__ gch, const __bf16* __restrict__ gcl,
                  const float* __restrict__ csq,
                  float* __restrict__ q_out, float* __restrict__ qpart) {
    // padded row stride 40 elems (80 B): 16-byte aligned b128 loads, no bank conflicts
    __shared__ __align__(16) __bf16 sxh[128 * 40];
    __shared__ __align__(16) __bf16 sxl[128 * 40];
    __shared__ __align__(16) __bf16 sch[256 * 40];
    __shared__ __align__(16) __bf16 scl[256 * 40];
    __shared__ float sxsq[128];
    __shared__ float scsq[256];

    const int tid  = threadIdx.x;
    const int row0 = blockIdx.x * 128;
    const int wave = tid >> 5;
    const int lane = tid & 31;
    const int half = lane >> 4;            // 0 | 1
    const int l16  = lane & 15;
    const int rw   = wave & 3;             // row group: rows rw*32 .. rw*32+31
    const int cw   = wave >> 2;            // col half:  cols cw*128 .. cw*128+127
    const int R0   = rw * 32;
    const int C0   = cw * 128;

    if (tid < 128) sxsq[tid] = 0.0f;
    scsq[tid] = csq[tid];

    v8f acc[2][8];
    #pragma unroll
    for (int g = 0; g < 2; ++g)
        #pragma unroll
        for (int t = 0; t < 8; ++t)
            #pragma unroll
            for (int e = 0; e < 8; ++e) acc[g][t][e] = 0.0f;

    float xsq_part = 0.0f;
    const int srow = tid >> 1;             // staging: 2 threads per row
    const int scg  = (tid & 1) * 16;       // 16 cols each

    for (int it = 0; it < 32; ++it) {
        const int dbase = it * 32;
        __syncthreads();                   // compute(prev) -> stage
        // --- async-stage cluster tile (bf16 in L2) straight into LDS ---
        {
            const uint32_t gb = (uint32_t)(((size_t)tid * DIM + dbase) * sizeof(__bf16));
            #pragma unroll
            for (int j = 0; j < 4; ++j) {
                async_copy_b128(lds_offset_of(&sch[tid * 40 + j * 8]), gch, gb + j * 16);
                async_copy_b128(lds_offset_of(&scl[tid * 40 + j * 8]), gcl, gb + j * 16);
            }
        }
        // --- stage x tile: fp32 -> bf16 hi/lo, accumulate ||x||^2 (overlaps async) ---
        {
            const float* xp = x + (size_t)(row0 + srow) * DIM + dbase + scg;
            if (it + 1 < 32) __builtin_prefetch(xp + 32, 0, 0);
            #pragma unroll
            for (int j = 0; j < 16; ++j) {
                float v  = xp[j];
                __bf16 h = (__bf16)v;
                sxh[srow * 40 + scg + j] = h;
                sxl[srow * 40 + scg + j] = (__bf16)(v - (float)h);
                xsq_part += v * v;
            }
        }
        wait_asynccnt0();                  // our async copies landed in LDS
        __syncthreads();                   // stage -> compute

        // A fragments (16x32 bf16: lanes<16 hold K 0-7,16-23; lanes>=16 hold K 8-15,24-31)
        union { v16bf v; v8bf h[2]; } Ah[2], Al[2];
        #pragma unroll
        for (int g = 0; g < 2; ++g) {
            const int aoff = (R0 + g * 16 + l16) * 40 + half * 8;
            Ah[g].h[0] = *(const v8bf*)&sxh[aoff];
            Ah[g].h[1] = *(const v8bf*)&sxh[aoff + 16];
            Al[g].h[0] = *(const v8bf*)&sxl[aoff];
            Al[g].h[1] = *(const v8bf*)&sxl[aoff + 16];
        }
        #pragma unroll
        for (int t = 0; t < 8; ++t) {
            // B fragment (32x16: lanes<16 hold K 0-7,16-23; lanes>=16 hold K 8-15,24-31; col = lane&15)
            const int boff = (C0 + t * 16 + l16) * 40 + half * 8;
            union { v16bf v; v8bf h[2]; } Bh, Bl;
            Bh.h[0] = *(const v8bf*)&sch[boff];
            Bh.h[1] = *(const v8bf*)&sch[boff + 16];
            Bl.h[0] = *(const v8bf*)&scl[boff];
            Bl.h[1] = *(const v8bf*)&scl[boff + 16];
            #pragma unroll
            for (int g = 0; g < 2; ++g) {  // B reused across both row groups
                acc[g][t] = __builtin_amdgcn_wmma_f32_16x16x32_bf16(false, Ah[g].v, false, Bh.v,
                                                                    (short)0, acc[g][t], false, false);
                acc[g][t] = __builtin_amdgcn_wmma_f32_16x16x32_bf16(false, Ah[g].v, false, Bl.v,
                                                                    (short)0, acc[g][t], false, false);
                acc[g][t] = __builtin_amdgcn_wmma_f32_16x16x32_bf16(false, Al[g].v, false, Bh.v,
                                                                    (short)0, acc[g][t], false, false);
            }
        }
    }

    // ||x||^2 per row (2 commutative adds per row -> deterministic)
    atomicAdd(&sxsq[srow], xsq_part);
    __syncthreads();                       // also: staging LDS now safe to reuse

    float xs[2][8];
    #pragma unroll
    for (int g = 0; g < 2; ++g)
        #pragma unroll
        for (int j = 0; j < 8; ++j) xs[g][j] = sxsq[R0 + g * 16 + j + 8 * half];

    // q_unnorm = 1/(1+dist), overwrite accumulators
    #pragma unroll
    for (int t = 0; t < 8; ++t) {
        const float cs = scsq[C0 + t * 16 + l16];
        #pragma unroll
        for (int g = 0; g < 2; ++g)
            #pragma unroll
            for (int j = 0; j < 8; ++j) {
                float dist    = xs[g][j] + cs - 2.0f * acc[g][t][j];
                acc[g][t][j]  = 1.0f / (1.0f + dist);
            }
    }
    // partial row sums over this wave's 128 cols; combine col-halves via LDS
    float* srs = (float*)scl;              // reuse staging LDS: [128 rows][2 col-halves]
    #pragma unroll
    for (int g = 0; g < 2; ++g)
        #pragma unroll
        for (int j = 0; j < 8; ++j) {
            float s = 0.0f;
            #pragma unroll
            for (int t = 0; t < 8; ++t) s += acc[g][t][j];
            #pragma unroll
            for (int m = 1; m < 16; m <<= 1) s += __shfl_xor(s, m, 32);
            if (l16 == 0) srs[(R0 + g * 16 + j + 8 * half) * 2 + cw] = s;
        }
    __syncthreads();
    float rs[2][8];
    #pragma unroll
    for (int g = 0; g < 2; ++g)
        #pragma unroll
        for (int j = 0; j < 8; ++j) {
            const int row = R0 + g * 16 + j + 8 * half;
            rs[g][j] = 1.0f / (srs[row * 2 + 0] + srs[row * 2 + 1]);   // fixed order
        }
    // normalize, emit q, per-row-group column partials (deterministic WG reduce)
    float* swp = (float*)sch;              // reuse staging LDS: [4 row groups][256 cols]
    #pragma unroll
    for (int t = 0; t < 8; ++t) {
        float colp = 0.0f;
        #pragma unroll
        for (int g = 0; g < 2; ++g)
            #pragma unroll
            for (int j = 0; j < 8; ++j) {
                float qn = acc[g][t][j] * rs[g][j];
                colp += qn;
                const int grow = row0 + R0 + g * 16 + j + 8 * half;
                q_out[(size_t)grow * KC + C0 + t * 16 + l16] = qn;
            }
        colp += __shfl_xor(colp, 16, 32);  // merge halves -> sum over 32 rows
        if (half == 0) swp[rw * 256 + C0 + t * 16 + l16] = colp;
    }
    __syncthreads();
    float s = 0.0f;
    #pragma unroll
    for (int g = 0; g < 4; ++g) s += swp[g * 256 + tid];
    qpart[(size_t)blockIdx.x * KC + tid] = s;
}

// fixed-order reduction of 512 per-block column partials -> q_sum[K]
__global__ void reduce_qsum_kernel(const float* __restrict__ qpart, float* __restrict__ qsum) {
    const int k = threadIdx.x;
    float s = 0.0f;
    for (int b = 0; b < 512; ++b) s += qpart[(size_t)b * KC + k];
    qsum[k] = s;
}

// ---------------------------------------------------------------------------
// Pass 2: wave per row. q^2/q_sum, normalize, argmax (first-index tie-break),
// threshold, scatter v*x into numerator / v into denominator.
// ---------------------------------------------------------------------------
__global__ __launch_bounds__(256)
void pass2_kernel(const float* __restrict__ q, const float* __restrict__ x,
                  const float* __restrict__ qsum,
                  float* __restrict__ num, float* __restrict__ den) {
    __shared__ float sq[256];
    sq[threadIdx.x] = qsum[threadIdx.x];
    __syncthreads();

    const int wave = threadIdx.x >> 5;
    const int lane = threadIdx.x & 31;
    const int n    = blockIdx.x * 8 + wave;
    const float* qr = q + (size_t)n * KC;

    float s2 = 0.0f, bmax = -1.0f;
    int bidx = 0;
    #pragma unroll
    for (int i = 0; i < 8; ++i) {
        const int k = lane + i * 32;
        float v  = qr[k];
        float q2 = v * v / sq[k];
        s2 += q2;
        if (q2 > bmax) { bmax = q2; bidx = k; }   // ascending k -> first max kept
    }
    #pragma unroll
    for (int m = 16; m > 0; m >>= 1) {
        float ov = __shfl_xor(bmax, m, 32);
        int   oi = __shfl_xor(bidx, m, 32);
        if (ov > bmax || (ov == bmax && oi < bidx)) { bmax = ov; bidx = oi; }
        s2 += __shfl_xor(s2, m, 32);
    }
    const float v = bmax / s2;                    // normalized sharpened value at argmax
    const float out_v = (v > THRESH) ? v : 0.0f;  // relu(v-T)+sign*T == (v>T ? v : 0)
    if (out_v > 0.0f) {
        if (lane == 0) atomicAdd(&den[bidx], out_v);
        const float* xr = x   + (size_t)n    * DIM;
        float*       nr = num + (size_t)bidx * DIM;
        #pragma unroll 4
        for (int d = lane; d < DIM; d += 32) atomicAdd(&nr[d], out_v * xr[d]);
    }
}

__global__ void finalize_kernel(const float* __restrict__ num, const float* __restrict__ den,
                                float* __restrict__ out_clusters) {
    const int i = blockIdx.x * 256 + threadIdx.x;   // grid 1024 -> 256*1024
    const int k = i >> 10;
    out_clusters[i] = num[i] / (den[k] + EPS_F);
}

// ---------------------------------------------------------------------------
extern "C" void kernel_launch(void* const* d_in, const int* in_sizes, int n_in,
                              void* d_out, int out_size, void* d_ws, size_t ws_size,
                              hipStream_t stream) {
    const float* x        = (const float*)d_in[0];   // [N, D]
    const float* clusters = (const float*)d_in[1];   // [K, D]

    float* q_out        = (float*)d_out;                               // [N, K]
    float* out_clusters = (float*)d_out + (size_t)N_ROWS * KC;         // [K, D]

    char* ws = (char*)d_ws;
    float*  csq   = (float*)(ws + WS_CSQ);
    float*  qsum  = (float*)(ws + WS_QSUM);
    float*  den   = (float*)(ws + WS_DEN);
    float*  num   = (float*)(ws + WS_NUM);
    __bf16* ch    = (__bf16*)(ws + WS_CH);
    __bf16* cl    = (__bf16*)(ws + WS_CL);
    float*  qpart = (float*)(ws + WS_QPART);

    zero_ws_kernel      <<<1024, 256, 0, stream>>>(num, den);
    prep_clusters_kernel<<< 256, 256, 0, stream>>>(clusters, ch, cl, csq);
    pass1_kernel        <<< 512, 256, 0, stream>>>(x, ch, cl, csq, q_out, qpart);
    reduce_qsum_kernel  <<<   1, 256, 0, stream>>>(qpart, qsum);
    pass2_kernel        <<<8192, 256, 0, stream>>>(q_out, x, qsum, num, den);
    finalize_kernel     <<<1024, 256, 0, stream>>>(num, den, out_clusters);
}
